// matrix_nufft_68032281968973
// MI455X (gfx1250) — matrix operand order repaired and checked by reading, not compiled, not run on hardware
//
#include <hip/hip_runtime.h>

// CDNA5 / gfx1250 wave32 WMMA types
typedef __attribute__((ext_vector_type(16))) _Float16 v16h;
typedef __attribute__((ext_vector_type(8)))  float    v8f;
typedef __attribute__((ext_vector_type(2)))  __fp16   fp16x2;  // matches cvt_pkrtz return

#define IM   128      // image side (R = IM*IM)
#define NC   8        // coils
#define NK   8192     // trajectory points
#define KT   16       // k-tile per workgroup (= WMMA N)
#define AXS  129      // padded LDS row stride for Ax (kills 16-way bank conflicts)

// v_cvt_pk_rtz_f16_f32: two f32 -> packed 2xf16 in one VALU op
__device__ __forceinline__ unsigned pk2(float a, float b) {
    union { fp16x2 h; unsigned u; } u;
    u.h = __builtin_amdgcn_cvt_pkrtz(a, b);
    return u.u;
}

// ---------------------------------------------------------------------------
// Prepass: convert the static image into WMMA A-fragment-ordered packed f16.
// Fragment index (halves): ((((c*8 + mt)*4 + cc)*32 + lane)*16 + elem)
//   lane<16 holds row M=lane,   K runs {0..7, 16..23}
//   lane>=16 holds row M=lane-16, K runs {8..15, 24..31}
//   elem e -> K = e (e<8) else e+8
// Stored as dwords (2 halves each): elem2 = e/2, K = 2*elem2 (+8 if elem2>=4)
// ---------------------------------------------------------------------------
__global__ __launch_bounds__(128)
void pack_img_kernel(const float* __restrict__ imgRe,
                     const float* __restrict__ imgIm,
                     unsigned* __restrict__ wsRe,
                     unsigned* __restrict__ wsIm)
{
    const int total = NC * IM * IM / 2;   // 65536 dwords per array
    for (int d = blockIdx.x * blockDim.x + threadIdx.x; d < total;
         d += gridDim.x * blockDim.x) {
        const int elem2 = d & 7;
        const int lane  = (d >> 3) & 31;
        const int cc    = (d >> 8) & 3;
        const int mt    = (d >> 10) & 7;
        const int c     = d >> 13;
        const int n     = lane & 15;
        const int iyOff = (lane >> 4) ? 8 : 0;
        const int K     = (elem2 < 4) ? (2 * elem2) : (2 * elem2 + 8);
        const int iy    = cc * 32 + iyOff + K;
        const int row   = mt * 16 + n;
        const int src   = (c * IM + row) * IM + iy;
        wsRe[d] = pk2(imgRe[src], imgRe[src + 1]);
        wsIm[d] = pk2(imgIm[src], imgIm[src + 1]);
    }
}

// ---------------------------------------------------------------------------
// Shared twiddle setup: separable exp(-2pi j k.r) = Ax[k,ix] * Ay[k,iy].
// v_sin_f32/v_cos_f32 natively compute sin/cos(2*pi*x); phase is already in
// revolutions (|p| <= 32, inside HW domain) -> no range reduction.
// ---------------------------------------------------------------------------
__device__ __forceinline__ void setup_twiddles(
    const float* __restrict__ trj, int k0, int tid,
    float* sAxRe, float* sAxIm, _Float16* sAyRe, _Float16* sAyIm)
{
    for (int kl = 0; kl < KT; ++kl) {
        const float kx = trj[(k0 + kl) * 2 + 0];   // uniform -> SMEM load
        const float ky = trj[(k0 + kl) * 2 + 1];
        {   // Ax[kl, ix] (f32, epilogue)
            const int ix = tid;
            const float p = kx * (float)(ix - IM / 2) * (1.0f / IM);
            sAxRe[kl * AXS + ix] =  __builtin_amdgcn_cosf(p);
            sAxIm[kl * AXS + ix] = -__builtin_amdgcn_sinf(p);
        }
        {   // Ay[kl, iy] (f16, WMMA B-fragment order)
            const int iy = tid;
            const float p = ky * (float)(iy - IM / 2) * (1.0f / IM);
            const float re =  __builtin_amdgcn_cosf(p);
            const float im = -__builtin_amdgcn_sinf(p);
            const int cc  = iy >> 5;
            const int e   = iy & 31;
            const int bl  = ((e >> 3) & 1) ? (16 + kl) : kl;
            const int off = ((cc * 32 + bl) << 4) + ((e & 7) | ((e >> 4) << 3));
            sAyRe[off] = (_Float16)re;
            sAyIm[off] = (_Float16)im;
        }
    }
}

// ---------------------------------------------------------------------------
// Fast path: A-fragments pre-packed in ws (f16), no cvt in the hot loop.
// ---------------------------------------------------------------------------
__global__ __launch_bounds__(128)
void nufft_fwd_fast(const _Float16* __restrict__ wsRe,
                    const _Float16* __restrict__ wsIm,
                    const float* __restrict__ trj,
                    float* __restrict__ out)
{
    __shared__ float sAxRe[KT * AXS];
    __shared__ float sAxIm[KT * AXS];
    __shared__ __align__(32) _Float16 sAyRe[4 * 32 * 16];
    __shared__ __align__(32) _Float16 sAyIm[4 * 32 * 16];

    const int tid  = threadIdx.x;
    const int lane = tid & 31;
    const int wave = tid >> 5;
    const int k0   = blockIdx.x * KT;

    setup_twiddles(trj, k0, tid, sAxRe, sAxIm, sAyRe, sAyIm);
    __syncthreads();

    const int n    = lane & 15;
    const int hsel = lane >> 4;

    for (int cidx = 0; cidx < 2; ++cidx) {
        const int c = wave * 2 + cidx;
        float outRe = 0.f, outIm = 0.f;

        for (int mt = 0; mt < 8; ++mt) {
            v8f accRe = {0.f,0.f,0.f,0.f,0.f,0.f,0.f,0.f};
            v8f accIm = {0.f,0.f,0.f,0.f,0.f,0.f,0.f,0.f};
            const int fwave = (((c * 8 + mt) * 4) * 32 + lane) << 4; // halves

            #pragma unroll
            for (int cc = 0; cc < 4; ++cc) {
                const int fbase = fwave + (cc << 9);         // +cc*32*16
                const v16h frv = *(const v16h*)(wsRe + fbase);
                union { v16h v; unsigned u[8]; } fi, fin;
                fi.v = *(const v16h*)(wsIm + fbase);
                if (cc < 3) {                                 // near-cache prefetch
                    __builtin_prefetch(wsRe + fbase + 512, 0, 3);
                    __builtin_prefetch(wsIm + fbase + 512, 0, 3);
                }
                #pragma unroll
                for (int j = 0; j < 8; ++j)                   // -imag: sign-bit flip
                    fin.u[j] = fi.u[j] ^ 0x80008000u;

                const v16h gr = *(const v16h*)&sAyRe[(cc * 32 + lane) << 4];
                const v16h gi = *(const v16h*)&sAyIm[(cc * 32 + lane) << 4];

                accRe = __builtin_amdgcn_wmma_f32_16x16x32_f16(false, frv,   false, gr, (short)0, accRe, false, false);
                accRe = __builtin_amdgcn_wmma_f32_16x16x32_f16(false, fin.v, false, gi, (short)0, accRe, false, false);
                accIm = __builtin_amdgcn_wmma_f32_16x16x32_f16(false, frv,   false, gi, (short)0, accIm, false, false);
                accIm = __builtin_amdgcn_wmma_f32_16x16x32_f16(false, fi.v,  false, gr, (short)0, accIm, false, false);
            }

            union { v8f v; float f[8]; } tre, tim;
            tre.v = accRe; tim.v = accIm;
            const int mBase = mt * 16 + (hsel ? 8 : 0);
            #pragma unroll
            for (int j = 0; j < 8; ++j) {
                const float axr = sAxRe[n * AXS + mBase + j];
                const float axi = sAxIm[n * AXS + mBase + j];
                outRe += axr * tre.f[j] - axi * tim.f[j];
                outIm += axr * tim.f[j] + axi * tre.f[j];
            }
        }

        outRe += __shfl_down(outRe, 16);
        outIm += __shfl_down(outIm, 16);
        if (lane < 16) {
            const int k = k0 + n;
            out[(c * NK + k) * 2 + 0] = outRe;
            out[(c * NK + k) * 2 + 1] = outIm;
        }
    }
}

// ---------------------------------------------------------------------------
// Fallback (ws too small): convert f32 image inline (previous round's kernel).
// ---------------------------------------------------------------------------
__global__ __launch_bounds__(128)
void nufft_fwd_fallback(const float* __restrict__ imgRe,
                        const float* __restrict__ imgIm,
                        const float* __restrict__ trj,
                        float* __restrict__ out)
{
    __shared__ float sAxRe[KT * AXS];
    __shared__ float sAxIm[KT * AXS];
    __shared__ __align__(32) _Float16 sAyRe[4 * 32 * 16];
    __shared__ __align__(32) _Float16 sAyIm[4 * 32 * 16];

    const int tid  = threadIdx.x;
    const int lane = tid & 31;
    const int wave = tid >> 5;
    const int k0   = blockIdx.x * KT;

    setup_twiddles(trj, k0, tid, sAxRe, sAxIm, sAyRe, sAyIm);
    __syncthreads();

    const int n     = lane & 15;
    const int hsel  = lane >> 4;
    const int iyOff = hsel ? 8 : 0;

    for (int cidx = 0; cidx < 2; ++cidx) {
        const int c = wave * 2 + cidx;
        float outRe = 0.f, outIm = 0.f;

        for (int mt = 0; mt < 8; ++mt) {
            const int mrow = mt * 16 + n;
            const float* bre = imgRe + (c * IM + mrow) * IM + iyOff;
            const float* bim = imgIm + (c * IM + mrow) * IM + iyOff;

            v8f accRe = {0.f,0.f,0.f,0.f,0.f,0.f,0.f,0.f};
            v8f accIm = {0.f,0.f,0.f,0.f,0.f,0.f,0.f,0.f};

            #pragma unroll
            for (int cc = 0; cc < 4; ++cc) {
                const float4* pr = (const float4*)(bre + cc * 32);
                const float4* pi = (const float4*)(bim + cc * 32);
                const float4 r0 = pr[0], r1 = pr[1], r2 = pr[4], r3 = pr[5];
                const float4 i0 = pi[0], i1 = pi[1], i2 = pi[4], i3 = pi[5];
                if (cc < 3) {
                    __builtin_prefetch(bre + (cc + 1) * 32, 0, 3);
                    __builtin_prefetch(bim + (cc + 1) * 32, 0, 3);
                }

                union { v16h v; unsigned u[8]; } fr, fi, fin;
                fr.u[0] = pk2(r0.x, r0.y); fr.u[1] = pk2(r0.z, r0.w);
                fr.u[2] = pk2(r1.x, r1.y); fr.u[3] = pk2(r1.z, r1.w);
                fr.u[4] = pk2(r2.x, r2.y); fr.u[5] = pk2(r2.z, r2.w);
                fr.u[6] = pk2(r3.x, r3.y); fr.u[7] = pk2(r3.z, r3.w);
                fi.u[0] = pk2(i0.x, i0.y); fi.u[1] = pk2(i0.z, i0.w);
                fi.u[2] = pk2(i1.x, i1.y); fi.u[3] = pk2(i1.z, i1.w);
                fi.u[4] = pk2(i2.x, i2.y); fi.u[5] = pk2(i2.z, i2.w);
                fi.u[6] = pk2(i3.x, i3.y); fi.u[7] = pk2(i3.z, i3.w);
                #pragma unroll
                for (int j = 0; j < 8; ++j)
                    fin.u[j] = fi.u[j] ^ 0x80008000u;

                const v16h gr = *(const v16h*)&sAyRe[(cc * 32 + lane) << 4];
                const v16h gi = *(const v16h*)&sAyIm[(cc * 32 + lane) << 4];

                accRe = __builtin_amdgcn_wmma_f32_16x16x32_f16(false, fr.v,  false, gr, (short)0, accRe, false, false);
                accRe = __builtin_amdgcn_wmma_f32_16x16x32_f16(false, fin.v, false, gi, (short)0, accRe, false, false);
                accIm = __builtin_amdgcn_wmma_f32_16x16x32_f16(false, fr.v,  false, gi, (short)0, accIm, false, false);
                accIm = __builtin_amdgcn_wmma_f32_16x16x32_f16(false, fi.v,  false, gr, (short)0, accIm, false, false);
            }

            union { v8f v; float f[8]; } tre, tim;
            tre.v = accRe; tim.v = accIm;
            const int mBase = mt * 16 + (hsel ? 8 : 0);
            #pragma unroll
            for (int j = 0; j < 8; ++j) {
                const float axr = sAxRe[n * AXS + mBase + j];
                const float axi = sAxIm[n * AXS + mBase + j];
                outRe += axr * tre.f[j] - axi * tim.f[j];
                outIm += axr * tim.f[j] + axi * tre.f[j];
            }
        }

        outRe += __shfl_down(outRe, 16);
        outIm += __shfl_down(outIm, 16);
        if (lane < 16) {
            const int k = k0 + n;
            out[(c * NK + k) * 2 + 0] = outRe;
            out[(c * NK + k) * 2 + 1] = outIm;
        }
    }
}

extern "C" void kernel_launch(void* const* d_in, const int* in_sizes, int n_in,
                              void* d_out, int out_size, void* d_ws, size_t ws_size,
                              hipStream_t stream) {
    (void)in_sizes; (void)n_in; (void)out_size;
    const float* imgRe = (const float*)d_in[0];
    const float* imgIm = (const float*)d_in[1];
    const float* trj   = (const float*)d_in[2];
    float* out = (float*)d_out;

    const size_t needWs = (size_t)2 * NC * IM * IM * sizeof(_Float16); // 512 KB
    if (d_ws != nullptr && ws_size >= needWs) {
        _Float16* wsRe = (_Float16*)d_ws;
        _Float16* wsIm = wsRe + (size_t)NC * IM * IM;
        pack_img_kernel<<<dim3(256), dim3(128), 0, stream>>>(
            imgRe, imgIm, (unsigned*)wsRe, (unsigned*)wsIm);
        nufft_fwd_fast<<<dim3(NK / KT), dim3(128), 0, stream>>>(wsRe, wsIm, trj, out);
    } else {
        nufft_fwd_fallback<<<dim3(NK / KT), dim3(128), 0, stream>>>(imgRe, imgIm, trj, out);
    }
}
